// MultiHeadAttention_29154238005976
// MI455X (gfx1250) — compile-verified, matrix operand order checked by reading, not run on hardware
//
#include <hip/hip_runtime.h>
#include <hip/hip_bf16.h>

// ---------------------------------------------------------------------------
// Conv-augmented MHA for MI455X (gfx1250, wave32, WMMA bf16).
// B=4, L=1024, D_MODEL=1024, H=16, DK=64, FMAX=4, FILTER_LENGTHS={2,4}.
// All GEMM-shaped work runs on v_wmma_f32_16x16x32_bf16 with LDS-staged
// bf16 tiles; operands are contiguous 32B-per-lane ds_load_b128 reads.
// All hot-path indexing is 32-bit (buffers < 16MB) so loads use
// saddr + 32-bit voffset addressing instead of 64-bit VALU address chains.
// ---------------------------------------------------------------------------

typedef __attribute__((ext_vector_type(16))) __bf16 bf16x16;
typedef __attribute__((ext_vector_type(8)))  float  f32x8;

static __device__ inline f32x8 wmma_bf16(bf16x16 a, bf16x16 b, f32x8 c) {
  return __builtin_amdgcn_wmma_f32_16x16x32_bf16(
      /*neg_a=*/false, a, /*neg_b=*/false, b,
      /*c_mod=*/(short)0, c, /*reuse_a=*/false, /*reuse_b=*/false);
}

// Contiguous 16 bf16 (32 bytes) from LDS, 16B-aligned -> 2x ds_load_b128.
static __device__ inline bf16x16 lds_load_row(const __bf16* p) {
  union { bf16x16 v; uint4 q[2]; } u;
  u.q[0] = *(const uint4*)(p);
  u.q[1] = *(const uint4*)(p + 8);
  return u.v;
}

// 16 consecutive fp32 from global (16B-aligned) -> bf16x16.
static __device__ inline bf16x16 load_row_f32(const float* p) {
  bf16x16 r;
#pragma unroll
  for (int i = 0; i < 4; ++i) {
    float4 f = *(const float4*)(p + i * 4);
    r[i * 4 + 0] = (__bf16)f.x;
    r[i * 4 + 1] = (__bf16)f.y;
    r[i * 4 + 2] = (__bf16)f.z;
    r[i * 4 + 3] = (__bf16)f.w;
  }
  return r;
}

// Pack 4 floats -> 4 bf16 in one 8B value.
static __device__ inline uint2 pack4_bf16(float a, float b, float c, float d) {
  union { __bf16 h[4]; uint2 u; } pk;
  pk.h[0] = (__bf16)a; pk.h[1] = (__bf16)b;
  pk.h[2] = (__bf16)c; pk.h[3] = (__bf16)d;
  return pk.u;
}

// ---------------------------------------------------------------------------
// Tap mask: f_s = {2,4}; w_f = f_s * w; flen = FILTER_LENGTHS[argmax(w_f)];
// mask[f] = (f < flen). jnp.argmax takes the FIRST max on ties.
// ---------------------------------------------------------------------------
__global__ void mask_kernel(const float* __restrict__ w, float* __restrict__ mask) {
  if (threadIdx.x == 0) {
    float w0 = 2.0f * w[0];
    float w1 = 4.0f * w[1];
    int flen = (w1 > w0) ? 4 : 2;
    for (int f = 0; f < 4; ++f) mask[f] = (f < flen) ? 1.0f : 0.0f;
  }
}

// ---------------------------------------------------------------------------
// Tile geometry: block 128x128, K-step 32, 256 threads
// (8 waves: 4 over M x 2 over N -> each wave 2x4 = 8 wmma tiles).
// ASTR/BTSTR = 40 bf16 (80B rows, 16B aligned).
// ---------------------------------------------------------------------------
#define BM 128
#define BN 128
#define BK 32
#define ASTR 40
#define BTSTR 40

// Fill one K-step of tiles: As[r][c] = A[m0+r][kb+c] (128x32),
// Bst[c][r] = W[kb+r][n0+c] transposed (128 cols x 32 k).
static __device__ inline void fill_tiles(const float* __restrict__ A,
                                         const float* __restrict__ W,
                                         __bf16* __restrict__ As,
                                         __bf16* __restrict__ Bst,
                                         int m0, int n0, int kb, int K, int N,
                                         int tid) {
#pragma unroll
  for (int i = 0; i < 4; ++i) {               // A: 1024 float4 slots
    int e = tid + i * 256;
    int r = e >> 3, c4 = (e & 7) * 4;
    float4 f = *(const float4*)&A[(unsigned)((m0 + r) * K + kb + c4)];
    *(uint2*)&As[r * ASTR + c4] = pack4_bf16(f.x, f.y, f.z, f.w);
  }
#pragma unroll
  for (int i = 0; i < 4; ++i) {               // B: 1024 float4 slots
    int e = tid + i * 256;
    int r = e >> 5, c4 = (e & 31) * 4;
    float4 f = *(const float4*)&W[(unsigned)((kb + r) * N + n0 + c4)];
    Bst[(c4 + 0) * BTSTR + r] = (__bf16)f.x;
    Bst[(c4 + 1) * BTSTR + r] = (__bf16)f.y;
    Bst[(c4 + 2) * BTSTR + r] = (__bf16)f.z;
    Bst[(c4 + 3) * BTSTR + r] = (__bf16)f.w;
  }
}

// ---------------------------------------------------------------------------
// Generic GEMM: C(MxN) = A(MxK) @ W(KxN) + bias, double-buffered LDS.
// MODE 0: C[m*N+n]
// MODE 1: qkv transpose -> C[((b*16+h)*1024+l)*64+dk], m=b*1024+l, n=h*64+dk
// ---------------------------------------------------------------------------
template <int MODE>
__global__ __launch_bounds__(256) void gemm_bias_kernel(
    const float* __restrict__ A, const float* __restrict__ W,
    const float* __restrict__ bias, float* __restrict__ C,
    int M, int N, int K) {
  __shared__ __align__(16) __bf16 As[2][BM * ASTR];
  __shared__ __align__(16) __bf16 Bst[2][BN * BTSTR];

  int tid = threadIdx.x;
  int wave = tid >> 5;
  int lane = tid & 31;
  int lr = lane & 15, kh16 = (lane >> 4) * 16;
  int m0 = blockIdx.y * BM;
  int n0 = blockIdx.x * BN;
  int wm = (wave >> 1) * 32;   // 4 waves over M, 32 rows each
  int wn = (wave & 1) * 64;    // 2 waves over N, 64 cols each

  f32x8 acc[2][4] = {};

  fill_tiles(A, W, As[0], Bst[0], m0, n0, 0, K, N, tid);
  __syncthreads();

  for (int kb = 0; kb < K; kb += BK) {
    int cur = (kb >> 5) & 1, nxt = cur ^ 1;
    if (kb + BK < K)
      fill_tiles(A, W, As[nxt], Bst[nxt], m0, n0, kb + BK, K, N, tid);

    bf16x16 a[2], bb[4];
#pragma unroll
    for (int im = 0; im < 2; ++im)
      a[im] = lds_load_row(&As[cur][(wm + im * 16 + lr) * ASTR + kh16]);
#pragma unroll
    for (int j = 0; j < 4; ++j)
      bb[j] = lds_load_row(&Bst[cur][(wn + j * 16 + lr) * BTSTR + kh16]);
#pragma unroll
    for (int im = 0; im < 2; ++im)
#pragma unroll
      for (int jn = 0; jn < 4; ++jn)
        acc[im][jn] = wmma_bf16(a[im], bb[jn], acc[im][jn]);
    __syncthreads();
  }

  // C/D layout: VGPR i -> row i + 8*(lane/16); column = lane%16.
  int rh = (lane >> 4) * 8;
#pragma unroll
  for (int im = 0; im < 2; ++im)
#pragma unroll
    for (int jn = 0; jn < 4; ++jn)
#pragma unroll
      for (int i = 0; i < 8; ++i) {
        int m = m0 + wm + im * 16 + rh + i;
        int n = n0 + wn + jn * 16 + lr;
        float v = acc[im][jn][i] + bias[n];
        if constexpr (MODE == 0) {
          C[(unsigned)(m * N + n)] = v;
        } else {
          int b = m >> 10, l = m & 1023, h = n >> 6, dk = n & 63;
          C[(unsigned)((((b * 16 + h) << 10) + l) * 64 + dk)] = v;
        }
      }
}

// ---------------------------------------------------------------------------
// Conv + residual, matching the reference's flat reshape semantics.
// Per batch b, view x (flat 2^20) as X[c,t] (c=flat/1024, t=flat%1024):
//   Y[t,o] = sum_{c,f} mask[f] * X[c, t+f-2] * convw[c,o,f]   (zero pad in t)
//   xout_flat[t*1024+o] = Y[t,o] + xin_flat[t*1024+o]
// Fill mapping: each thread owns a contiguous 16-element run of one LDS row
// -> 4x packed 8B ds stores; global indices are 32-bit, loop-invariant bases.
// ---------------------------------------------------------------------------
static __device__ inline void conv_fill(const float* __restrict__ X,
                                        const float* __restrict__ convw,
                                        __bf16* __restrict__ As,
                                        __bf16* __restrict__ Bst,
                                        int t0, int n0, int kb, int f, int tid) {
  // A tile: row = tid&127 (t), cols (tid>>7)*16 .. +15 (c)
  {
    int row = tid & 127;
    int ch0 = (tid >> 7) * 16;
    int t = t0 + row + f - 2;                  // [-2, 1024]
    unsigned tcl = (unsigned)(t & 1023);
    float vm = ((unsigned)t < 1024u) ? 1.0f : 0.0f;
    __bf16* dst = &As[row * ASTR + ch0];
    unsigned base = (unsigned)(kb + ch0) * 1024u + tcl;
#pragma unroll
    for (int i4 = 0; i4 < 4; ++i4) {
      float v0 = X[base + (unsigned)(i4 * 4 + 0) * 1024u] * vm;
      float v1 = X[base + (unsigned)(i4 * 4 + 1) * 1024u] * vm;
      float v2 = X[base + (unsigned)(i4 * 4 + 2) * 1024u] * vm;
      float v3 = X[base + (unsigned)(i4 * 4 + 3) * 1024u] * vm;
      *(uint2*)(dst + i4 * 4) = pack4_bf16(v0, v1, v2, v3);
    }
  }
  // B tile: col = tid&127 (o), k rows (tid>>7)*16 .. +15 (c)
  {
    int c = tid & 127;
    int rh0 = (tid >> 7) * 16;
    __bf16* dst = &Bst[c * BTSTR + rh0];
    unsigned base = (unsigned)(kb + rh0) * 4096u + (unsigned)(n0 + c) * 4u
                  + (unsigned)f;
#pragma unroll
    for (int i4 = 0; i4 < 4; ++i4) {
      float v0 = convw[base + (unsigned)(i4 * 4 + 0) * 4096u];
      float v1 = convw[base + (unsigned)(i4 * 4 + 1) * 4096u];
      float v2 = convw[base + (unsigned)(i4 * 4 + 2) * 4096u];
      float v3 = convw[base + (unsigned)(i4 * 4 + 3) * 4096u];
      *(uint2*)(dst + i4 * 4) = pack4_bf16(v0, v1, v2, v3);
    }
  }
}

__global__ __launch_bounds__(256) void conv_residual_kernel(
    const float* __restrict__ xin, const float* __restrict__ convw,
    const float* __restrict__ mask, float* __restrict__ xout) {
  __shared__ __align__(16) __bf16 As[2][BM * ASTR];     // [t][c] 128x32
  __shared__ __align__(16) __bf16 Bst[2][BN * BTSTR];   // [o][c] 128x32

  int b = blockIdx.z;
  const float* X = xin + (size_t)b * (1024u * 1024u);
  int t0 = blockIdx.y * BM;
  int n0 = blockIdx.x * BN;
  int tid = threadIdx.x;
  int wave = tid >> 5;
  int lane = tid & 31;
  int lr = lane & 15, kh16 = (lane >> 4) * 16;
  int wm = (wave >> 1) * 32;
  int wn = (wave & 1) * 64;

  f32x8 acc[2][4] = {};

  for (int f = 0; f < 4; ++f) {
    float mk = mask[f];          // uniform across the grid
    if (mk == 0.0f) continue;

    conv_fill(X, convw, As[0], Bst[0], t0, n0, 0, f, tid);
    __syncthreads();
    for (int kb = 0; kb < 1024; kb += BK) {
      int cur = (kb >> 5) & 1, nxt = cur ^ 1;
      if (kb + BK < 1024)
        conv_fill(X, convw, As[nxt], Bst[nxt], t0, n0, kb + BK, f, tid);

      bf16x16 a[2], bb[4];
#pragma unroll
      for (int im = 0; im < 2; ++im)
        a[im] = lds_load_row(&As[cur][(wm + im * 16 + lr) * ASTR + kh16]);
#pragma unroll
      for (int j = 0; j < 4; ++j)
        bb[j] = lds_load_row(&Bst[cur][(wn + j * 16 + lr) * BTSTR + kh16]);
#pragma unroll
      for (int im = 0; im < 2; ++im)
#pragma unroll
        for (int jn = 0; jn < 4; ++jn)
          acc[im][jn] = wmma_bf16(a[im], bb[jn], acc[im][jn]);
      __syncthreads();
    }
  }

  int rh = (lane >> 4) * 8;
  unsigned bbase = (unsigned)b * (1024u * 1024u);
#pragma unroll
  for (int im = 0; im < 2; ++im)
#pragma unroll
    for (int jn = 0; jn < 4; ++jn)
#pragma unroll
      for (int i = 0; i < 8; ++i) {
        int t = t0 + wm + im * 16 + rh + i;
        int o = n0 + wn + jn * 16 + lr;
        unsigned idx = bbase + (unsigned)t * 1024u + (unsigned)o;
        xout[idx] = acc[im][jn][i] + xin[idx];
      }
}

// ---------------------------------------------------------------------------
// Attention: one WG per (b, h, 32 q-rows).
//  - Q block staged once in LDS; A operands hoisted to registers.
//  - Scores (32x1024) in LDS fp32 via WMMA QK^T (transpose trick).
//  - Row softmax with wave32 shuffles, float4 LDS reads, packed bf16 stores.
//  - P (bf16, LDS) @ V with WMMA; V staged in 128x64 LDS strips (reusing the
//    dead score buffer), transposed for aligned ds_load_b128 operands.
// Dynamic LDS: 132096 (S/Vst) + 66560 (P) + 128 (rowsum) + 4608 (Qs)
//            = 203392 bytes (< 320KB WGP LDS).
// ---------------------------------------------------------------------------
#define SS 1032          // score row stride (floats)
#define PS 1040          // P row stride (bf16), 16B-aligned rows
#define QSTR 72          // Q block row stride (bf16), 144B = 16B-aligned
#define VSTR 136         // V strip row stride (bf16), 272B = 16B-aligned

__global__ __launch_bounds__(256) void attention_kernel(
    const float* __restrict__ q, const float* __restrict__ k,
    const float* __restrict__ v, float* __restrict__ ctx) {
  extern __shared__ char smem_raw[];
  float*  S  = (float*)smem_raw;                 // 32 x SS fp32 (phase 1/2)
  __bf16* Vst = (__bf16*)smem_raw;               // 64 x VSTR bf16 (phase 3)
  __bf16* P  = (__bf16*)(S + 32 * SS);           // 32 x PS bf16
  float*  rowsum = (float*)(P + 32 * PS);        // 32 floats
  __bf16* Qs = (__bf16*)(rowsum + 32);           // 32 x QSTR bf16

  int b = blockIdx.z, h = blockIdx.y;
  int q0r = blockIdx.x * 32;
  const size_t head = ((size_t)(b * 16 + h)) << 10;   // *L
  const float* qh = q + head * 64;
  const float* kh = k + head * 64;
  const float* vh = v + head * 64;

  int tid = threadIdx.x, wave = tid >> 5, lane = tid & 31;
  int lr = lane & 15, kh16 = (lane >> 4) * 16;

  // ---- Phase 0: stage Q block (32x64) into LDS ----
#pragma unroll
  for (int i = 0; i < 2; ++i) {
    int e4 = tid + i * 256;                 // 512 float4 slots
    int row = e4 >> 4, c4 = (e4 & 15) * 4;
    float4 f = *(const float4*)&qh[(unsigned)((q0r + row) * 64 + c4)];
    *(uint2*)&Qs[row * QSTR + c4] = pack4_bf16(f.x, f.y, f.z, f.w);
  }
  __syncthreads();

  bf16x16 aq[2][2];                         // [mt][kb chunk]
#pragma unroll
  for (int mt = 0; mt < 2; ++mt)
#pragma unroll
    for (int kc = 0; kc < 2; ++kc)
      aq[mt][kc] = lds_load_row(&Qs[(mt * 16 + lr) * QSTR + kc * 32 + kh16]);

  // ---- Phase 1: S = (Q_blk K^T) * rsqrt(DK) ----
  for (int t = wave; t < 128; t += 8) {     // 2 m-tiles x 64 n-tiles
    int mt = t & 1, nt = t >> 1;
    f32x8 acc = {};
#pragma unroll
    for (int kc = 0; kc < 2; ++kc) {
      bf16x16 bb = load_row_f32(kh + (unsigned)((nt * 16 + lr) * 64 + kc * 32 + kh16));
      acc = wmma_bf16(aq[mt][kc], bb, acc); // K rows loaded A-style => K^T
    }
#pragma unroll
    for (int i = 0; i < 8; ++i)
      S[(mt * 16 + (lane >> 4) * 8 + i) * SS + nt * 16 + lr] = acc[i] * 0.125f;
  }
  __syncthreads();

  // ---- Phase 2: row softmax (8 threads per row, 128 elems each) ----
  {
    int row = tid >> 3, j = tid & 7;
    const float* srow = S + row * SS + j * 128;
    float mx = -1e30f;
#pragma unroll 8
    for (int e = 0; e < 128; e += 4) {
      float4 s4 = *(const float4*)(srow + e);
      mx = fmaxf(mx, fmaxf(fmaxf(s4.x, s4.y), fmaxf(s4.z, s4.w)));
    }
#pragma unroll
    for (int msk = 1; msk < 8; msk <<= 1) mx = fmaxf(mx, __shfl_xor(mx, msk, 32));
    float sum = 0.0f;
    __bf16* prow = P + row * PS + j * 128;
#pragma unroll 8
    for (int e = 0; e < 128; e += 4) {
      float4 s4 = *(const float4*)(srow + e);
      float p0 = __expf(s4.x - mx), p1 = __expf(s4.y - mx);
      float p2 = __expf(s4.z - mx), p3 = __expf(s4.w - mx);
      sum += (p0 + p1) + (p2 + p3);
      *(uint2*)(prow + e) = pack4_bf16(p0, p1, p2, p3);  // unnormalized
    }
#pragma unroll
    for (int msk = 1; msk < 8; msk <<= 1) sum += __shfl_xor(sum, msk, 32);
    if (j == 0) rowsum[row] = sum;
  }
  __syncthreads();

  // ---- Phase 3: ctx = (P @ V) / rowsum, V staged in LDS strips ----
  {
    int mt = wave >> 2, nt = wave & 3;      // 2 x 4 output tiles, 1 per wave
    f32x8 acc = {};
    for (int kb0 = 0; kb0 < 1024; kb0 += 128) {
      __syncthreads();                      // previous strip fully consumed
#pragma unroll
      for (int i = 0; i < 8; ++i) {         // stage V strip 128x64, transposed
        int e4 = tid + i * 256;             // 2048 float4 slots
        int r = e4 >> 4, c4 = (e4 & 15) * 4;
        float4 f = *(const float4*)&vh[(unsigned)((kb0 + r) * 64 + c4)];
        Vst[(c4 + 0) * VSTR + r] = (__bf16)f.x;
        Vst[(c4 + 1) * VSTR + r] = (__bf16)f.y;
        Vst[(c4 + 2) * VSTR + r] = (__bf16)f.z;
        Vst[(c4 + 3) * VSTR + r] = (__bf16)f.w;
      }
      __syncthreads();
#pragma unroll
      for (int kk = 0; kk < 128; kk += 32) {
        bf16x16 a  = lds_load_row(P + (mt * 16 + lr) * PS + kb0 + kk + kh16);
        bf16x16 bb = lds_load_row(&Vst[(nt * 16 + lr) * VSTR + kk + kh16]);
        acc = wmma_bf16(a, bb, acc);
      }
    }
#pragma unroll
    for (int i = 0; i < 8; ++i) {
      int row = mt * 16 + (lane >> 4) * 8 + i;
      int col = nt * 16 + lr;
      float val = acc[i] / rowsum[row];
      // ctx layout: (b, l, h*64+dk) so the final GEMM is a plain GEMM
      ctx[(unsigned)((b * 1024 + q0r + row) * 1024 + h * 64 + col)] = val;
    }
  }
}

// ---------------------------------------------------------------------------
// Host-side launch
// Inputs (setup_inputs order):
//  0:Q 1:K 2:V 3:attn_mask 4:WQ 5:bQ 6:WK 7:bK 8:WV 9:bV
//  10:Wfc 11:bfc 12:conv_q 13:conv_k 14:w
// ---------------------------------------------------------------------------
extern "C" void kernel_launch(void* const* d_in, const int* in_sizes, int n_in,
                              void* d_out, int out_size, void* d_ws, size_t ws_size,
                              hipStream_t stream) {
  const float* Q      = (const float*)d_in[0];
  const float* K      = (const float*)d_in[1];
  const float* V      = (const float*)d_in[2];
  const float* WQ     = (const float*)d_in[4];
  const float* bQ     = (const float*)d_in[5];
  const float* WK     = (const float*)d_in[6];
  const float* bK     = (const float*)d_in[7];
  const float* WV     = (const float*)d_in[8];
  const float* bV     = (const float*)d_in[9];
  const float* Wfc    = (const float*)d_in[10];
  const float* bfc    = (const float*)d_in[11];
  const float* conv_q = (const float*)d_in[12];
  const float* conv_k = (const float*)d_in[13];
  const float* w      = (const float*)d_in[14];
  float* out = (float*)d_out;

  const size_t SZ = (size_t)4 * 16 * 1024 * 64;   // 4,194,304 per buffer
  float* ws   = (float*)d_ws;
  float* q0   = ws;            // (b,h,l,dk)
  float* k0   = q0 + SZ;
  float* v0   = k0 + SZ;
  float* q1   = v0 + SZ;       // post conv+residual
  float* k1   = q1 + SZ;
  float* ctx  = k1 + SZ;       // (b,l,h*dk)
  float* mask = ctx + SZ;      // 4 floats

  mask_kernel<<<1, 32, 0, stream>>>(w, mask);

  dim3 gg(1024 / BN, 4096 / BM);   // (8, 32)
  gemm_bias_kernel<1><<<gg, 256, 0, stream>>>(Q, WQ, bQ, q0, 4096, 1024, 1024);
  gemm_bias_kernel<1><<<gg, 256, 0, stream>>>(K, WK, bK, k0, 4096, 1024, 1024);
  gemm_bias_kernel<1><<<gg, 256, 0, stream>>>(V, WV, bV, v0, 4096, 1024, 1024);

  dim3 gc(1024 / BN, 1024 / BM, 4);   // (8, 8, 4)
  conv_residual_kernel<<<gc, 256, 0, stream>>>(q0, conv_q, mask, q1);
  conv_residual_kernel<<<gc, 256, 0, stream>>>(k0, conv_k, mask, k1);

  dim3 ga(1024 / 32, 16, 4);          // (32, 16, 4)
  size_t smem = (size_t)32 * SS * 4 + (size_t)32 * PS * 2 + 32 * 4
              + (size_t)32 * QSTR * 2;
  attention_kernel<<<ga, 256, smem, stream>>>(q1, k1, v0, ctx);

  gemm_bias_kernel<0><<<gg, 256, 0, stream>>>(ctx, Wfc, bfc, out, 4096, 1024, 1024);
}
